// MultiHeadAttention_44049184588081
// MI455X (gfx1250) — compile-verified, matrix operand order checked by reading, not run on hardware
//
#include <hip/hip_runtime.h>
#include <hip/hip_bf16.h>

// ---------------------------------------------------------------------------
// MultiHeadAttention for MI455X (gfx1250, wave32, WMMA f16->f32).
// B=2, S=4096, D=512, H=8, DK=64.
// Pipeline:
//   1) proj_gemm x3 : Qh (pre-scaled by 0.125*log2e), Kh [B,H,S,64] f16 ;
//                     VhT [B,H,64,S] f16 (pre-transposed)
//   2) attn_kernel (flash-attention, S^T trick, exp2 domain): Xa [B,S,512] f16
//   3) proj_gemm<f16 A, f32 out>: d_out = Xa @ w_o
// Both kernels software-pipeline global->LDS staging through registers with
// double-buffered LDS (one barrier/iter). The softmax runs in the log2 domain
// (scores pre-scaled by 1/sqrt(dk)*log2e at projection time) so the hot loop
// uses the raw v_exp_f32 (exp2) with no per-element multiply or guard code.
// Workspace: 4 x 8MB = 32MB.
// ---------------------------------------------------------------------------

typedef _Float16 v16h __attribute__((ext_vector_type(16)));
typedef _Float16 h8   __attribute__((ext_vector_type(8)));
typedef _Float16 h4   __attribute__((ext_vector_type(4)));
typedef float    v8f  __attribute__((ext_vector_type(8)));

union F16x16 { v16h v; h8 h2[2]; _Float16 e[16]; };

template <bool C, class T, class F> struct cond_ { using type = T; };
template <class T, class F> struct cond_<false, T, F> { using type = F; };

#if defined(__has_builtin)
#if __has_builtin(__builtin_amdgcn_exp2f)
#define DEV_EXP2F(x) __builtin_amdgcn_exp2f(x)   // raw v_exp_f32
#endif
#endif
#ifndef DEV_EXP2F
#define DEV_EXP2F(x) exp2f(x)
#endif

__device__ __forceinline__ v8f wmma_f16(v16h a, v16h b, v8f c) {
  // D = A(16x32 f16) * B(32x16 f16) + C(16x16 f32)
  return __builtin_amdgcn_wmma_f32_16x16x32_f16(
      /*neg_a=*/false, a, /*neg_b=*/false, b,
      /*c_mod=*/(short)0, c, /*reuse_a=*/false, /*reuse_b=*/false);
}

namespace cfg {
constexpr int Bz = 2, S = 4096, D = 512, H = 8, DK = 64;
constexpr int M = Bz * S;          // 8192 GEMM rows
constexpr int N = D, K = D;        // 512, 512
constexpr int BM = 128, BN = 64, BK = 64;
// epilogue modes
constexpr int EPI_F32 = 0;         // fp32 [M,N] row-major (final projection)
constexpr int EPI_HS = 1;          // f16 head-split [B,H,S,DK]  (Q,K)
constexpr int EPI_HST = 2;         // f16 head-split transposed [B,H,DK,S] (V)
constexpr float LOG2E = 1.44269504088896340736f;
constexpr float QSCALE = 0.125f * LOG2E;       // 1/sqrt(DK) * log2(e)
constexpr float MASKED = -1e9f * LOG2E;        // -1e9 in the log2 domain
}

// ---------------------------------------------------------------------------
// GEMM: C[M,N] = oscale * (A[M,K] * W[K,N]).  A fp32 or fp16; W fp32->f16.
// 256 threads = 8 waves in a 4x2 grid; each wave computes 32x32 (2x2 WMMA).
// Double-buffered LDS + register staging for global->LDS pipelining.
// ---------------------------------------------------------------------------
template <bool A_F16, int EPI>
__global__ __launch_bounds__(256) void proj_gemm(
    const void* __restrict__ Ain, const float* __restrict__ W,
    void* __restrict__ Out, float oscale) {
  using namespace cfg;
  __shared__ _Float16 As[2][BM * BK];   // [m][k] row-major, k contiguous
  __shared__ _Float16 Bs[2][BN * BK];   // W^T tile: [n][k], k contiguous

  const int t = threadIdx.x;
  const int lane = t & 31, wid = t >> 5;
  const int wm = wid >> 1, wn = wid & 1;      // wave tile position
  const int g = (lane >> 4) & 1;              // half-wave group (K split)
  const int lr = lane & 15;
  const int m0 = blockIdx.x * BM;
  const int n0 = blockIdx.y * BN;

  using ARegT = typename cond_<A_F16, h8, float4>::type;
  constexpr int ANR = A_F16 ? 4 : 8;

  auto loadA = [&](int kt, ARegT (&ar)[ANR]) {
    if constexpr (A_F16) {
      const _Float16* A = (const _Float16*)Ain;
#pragma unroll
      for (int i = 0; i < ANR; ++i) {
        int flat = (t + i * 256) * 8;
        int r = flat >> 6, c = flat & 63;
        ar[i] = *(const h8*)&A[(size_t)(m0 + r) * K + kt + c];
      }
    } else {
      const float* A = (const float*)Ain;
#pragma unroll
      for (int i = 0; i < ANR; ++i) {
        int flat = (t + i * 256) * 4;
        int r = flat >> 6, c = flat & 63;
        ar[i] = *(const float4*)&A[(size_t)(m0 + r) * K + kt + c];
      }
    }
  };
  auto loadW = [&](int kt, float4 (&wr)[4]) {
#pragma unroll
    for (int i = 0; i < 4; ++i) {
      int flat = (t + i * 256) * 4;
      int kr = flat >> 6, nc = flat & 63;
      wr[i] = *(const float4*)&W[(size_t)(kt + kr) * N + n0 + nc];
    }
  };
  auto storeA = [&](int buf, ARegT (&ar)[ANR]) {
    if constexpr (A_F16) {
#pragma unroll
      for (int i = 0; i < ANR; ++i) {
        int flat = (t + i * 256) * 8;
        int r = flat >> 6, c = flat & 63;
        *(h8*)&As[buf][r * BK + c] = ar[i];
      }
    } else {
#pragma unroll
      for (int i = 0; i < ANR; ++i) {
        int flat = (t + i * 256) * 4;
        int r = flat >> 6, c = flat & 63;
        const float4 f = ar[i];
        h4 hv = {(_Float16)f.x, (_Float16)f.y, (_Float16)f.z, (_Float16)f.w};
        *(h4*)&As[buf][r * BK + c] = hv;
      }
    }
  };
  auto storeW = [&](int buf, float4 (&wr)[4]) {
#pragma unroll
    for (int i = 0; i < 4; ++i) {
      int flat = (t + i * 256) * 4;
      int kr = flat >> 6, nc = flat & 63;
      const float4 w = wr[i];
      Bs[buf][(nc + 0) * BK + kr] = (_Float16)w.x;
      Bs[buf][(nc + 1) * BK + kr] = (_Float16)w.y;
      Bs[buf][(nc + 2) * BK + kr] = (_Float16)w.z;
      Bs[buf][(nc + 3) * BK + kr] = (_Float16)w.w;
    }
  };

  v8f acc[2][2] = {};

  {  // prologue: stage tile 0
    ARegT ar[ANR]; float4 wr[4];
    loadA(0, ar); loadW(0, wr);
    storeA(0, ar); storeW(0, wr);
  }
  __syncthreads();

  constexpr int NKT = K / BK;   // 8
  int cur = 0;
  for (int it = 0; it < NKT; ++it) {
    ARegT ar[ANR]; float4 wr[4];
    const bool more = (it + 1 < NKT);
    if (more) { loadA((it + 1) * BK, ar); loadW((it + 1) * BK, wr); }

#pragma unroll
    for (int ks = 0; ks < 2; ++ks) {            // two 32-deep WMMA K steps
      F16x16 af[2], bf[2];
#pragma unroll
      for (int mb = 0; mb < 2; ++mb) {
        const _Float16* p =
            &As[cur][(wm * 32 + mb * 16 + lr) * BK + ks * 32 + 8 * g];
        af[mb].h2[0] = *(const h8*)p;
        af[mb].h2[1] = *(const h8*)(p + 16);
      }
#pragma unroll
      for (int nb = 0; nb < 2; ++nb) {
        const _Float16* p =
            &Bs[cur][(wn * 32 + nb * 16 + lr) * BK + ks * 32 + 8 * g];
        bf[nb].h2[0] = *(const h8*)p;
        bf[nb].h2[1] = *(const h8*)(p + 16);
      }
#pragma unroll
      for (int mb = 0; mb < 2; ++mb)
#pragma unroll
        for (int nb = 0; nb < 2; ++nb)
          acc[mb][nb] = wmma_f16(af[mb].v, bf[nb].v, acc[mb][nb]);
    }

    if (more) { storeA(cur ^ 1, ar); storeW(cur ^ 1, wr); }
    __syncthreads();
    cur ^= 1;
  }

  // ---- epilogue ----
#pragma unroll
  for (int mb = 0; mb < 2; ++mb)
#pragma unroll
    for (int nb = 0; nb < 2; ++nb) {
      const int n = wn * 32 + nb * 16 + lr;          // C layout: lane%16
      const int Cc = n0 + n;
      if constexpr (EPI == cfg::EPI_HST) {
        // [B,H,DK,S]: the 8 lane values vary m=s contiguously -> h8 store.
        const int R0 = m0 + wm * 32 + mb * 16 + 8 * g;   // i = 0 row
        const int bb = R0 >> 12, ss = R0 & (S - 1);
        const int hh = Cc >> 6, dk = Cc & 63;
        h8 ov;
#pragma unroll
        for (int i = 0; i < 8; ++i)
          ov[i] = (_Float16)(acc[mb][nb][i] * oscale);
        *(h8*)&((_Float16*)Out)[(((size_t)bb * H + hh) * DK + dk) * S + ss] = ov;
      } else {
#pragma unroll
        for (int i = 0; i < 8; ++i) {
          const int m = wm * 32 + mb * 16 + i + 8 * g; // C layout: vgpr+8g
          const int R = m0 + m;
          const float val = acc[mb][nb][i] * oscale;
          if constexpr (EPI == cfg::EPI_HS) {
            const int bb = R >> 12, ss = R & (S - 1);
            const int hh = Cc >> 6, dk = Cc & 63;
            ((_Float16*)Out)[(((size_t)bb * H + hh) * S + ss) * DK + dk] =
                (_Float16)val;
          } else {
            ((float*)Out)[(size_t)R * N + Cc] = val;
          }
        }
      }
    }
}

// ---------------------------------------------------------------------------
// Flash attention. Qh (pre-scaled by 0.125*log2e), Kh head-split [B,H,S,64];
// VhT pre-transposed [B,H,64,S]. Softmax in the exp2 (log2) domain.
// Block = 256 thr (8 waves) handles 128 query rows of one (b,h).
// Each wave owns 16 q-rows -> private online softmax (no cross-wave comm).
// Computes S^T = K*Q^T so the softmax k-reduction is across VGPRs in-lane
// (one shfl_xor(16) merges the two half-wave k-partitions), and the S^T
// C-layout IS the B-fragment layout for O^T = V^T * P^T (exp2 + cvt only).
// K/V^T blocks staged in double-buffered LDS via register pipelining.
// ---------------------------------------------------------------------------
__global__ __launch_bounds__(256) void attn_kernel(
    const _Float16* __restrict__ Qh, const _Float16* __restrict__ Kh,
    const _Float16* __restrict__ VhT, const int* __restrict__ mask,
    _Float16* __restrict__ Xout) {
  using namespace cfg;
  __shared__ _Float16 Ks[2][64 * DK];  // [key][dk], dk contiguous
  __shared__ _Float16 Vt[2][DK * 64];  // V^T: [dk][key], key contiguous

  const int t = threadIdx.x;
  const int lane = t & 31, wid = t >> 5;
  const int g = (lane >> 4) & 1, lr = lane & 15;
  const int h = blockIdx.x, qb = blockIdx.y, b = blockIdx.z;
  const int q0 = qb * 128 + wid * 16;
  const size_t bh  = ((size_t)b * H + h) * S;    // row base, [B,H,S,DK]
  const size_t bhT = ((size_t)b * H + h) * DK;   // row base, [B,H,DK,S]

  // staging geometry: 2 x 8 halves per thread per 64x64 block
  const int sk_key0 = (t * 8) >> 6, sk_dk = (t * 8) & 63;       // K: i = 0
  const int sk_key1 = ((t + 256) * 8) >> 6;                     // K: i = 1
  const int sv_dk0 = sk_key0, sv_key = sk_dk;                   // V^T roles swap
  const int sv_dk1 = sk_key1;

  auto loadKV = [&](int kb, h8 (&kr)[2], h8 (&vr)[2]) {
    kr[0] = *(const h8*)&Kh[(bh + kb + sk_key0) * DK + sk_dk];
    kr[1] = *(const h8*)&Kh[(bh + kb + sk_key1) * DK + sk_dk];
    vr[0] = *(const h8*)&VhT[(bhT + sv_dk0) * S + kb + sv_key];
    vr[1] = *(const h8*)&VhT[(bhT + sv_dk1) * S + kb + sv_key];
  };
  auto storeKV = [&](int buf, h8 (&kr)[2], h8 (&vr)[2]) {
    *(h8*)&Ks[buf][sk_key0 * DK + sk_dk] = kr[0];
    *(h8*)&Ks[buf][sk_key1 * DK + sk_dk] = kr[1];
    *(h8*)&Vt[buf][sv_dk0 * 64 + sv_key] = vr[0];
    *(h8*)&Vt[buf][sv_dk1 * 64 + sv_key] = vr[1];
  };

  // Q^T B-fragments (already carry the 1/sqrt(dk)*log2e factor).
  F16x16 qf[2];
  {
    const _Float16* qp = Qh + (bh + q0 + lr) * DK;
#pragma unroll
    for (int ks = 0; ks < 2; ++ks) {
      qf[ks].h2[0] = *(const h8*)(qp + ks * 32 + 8 * g);
      qf[ks].h2[1] = *(const h8*)(qp + ks * 32 + 16 + 8 * g);
    }
  }

  v8f oacc[4] = {};                       // O^T tiles: m=dk (4x16), n=q
  float mi = -3.0e38f, li = 0.0f;         // online softmax state (per q=lane)
  const int q_lane = q0 + lr;
  const int* mrow = mask + ((size_t)b * S + q_lane) * S;

  {  // prologue: stage block 0
    h8 kr[2], vr[2];
    loadKV(0, kr, vr);
    storeKV(0, kr, vr);
  }
  __syncthreads();

  constexpr int NIT = S / 64;   // 64 key blocks
  int cur = 0;
  for (int it = 0; it < NIT; ++it) {
    const int kb = it * 64;
    h8 krg[2], vrg[2];
    const bool more = (it + 1 < NIT);
    if (more) loadKV(kb + 64, krg, vrg);   // in flight during WMMAs below

    // ---- scores S^T (4 tiles of 16k x 16q) in log2 domain, mask ----
    float p[4][8];
    float pmax = -3.0e38f;
#pragma unroll
    for (int tt = 0; tt < 4; ++tt) {
      const _Float16* kp = &Ks[cur][(tt * 16 + lr) * DK];
      F16x16 kf0, kf1;
      kf0.h2[0] = *(const h8*)(kp + 8 * g);
      kf0.h2[1] = *(const h8*)(kp + 16 + 8 * g);
      kf1.h2[0] = *(const h8*)(kp + 32 + 8 * g);
      kf1.h2[1] = *(const h8*)(kp + 48 + 8 * g);
      v8f sacc = {};
      sacc = wmma_f16(kf0.v, qf[0].v, sacc);
      sacc = wmma_f16(kf1.v, qf[1].v, sacc);
      const int4* mp = (const int4*)(mrow + kb + tt * 16 + 8 * g);
      const int4 mlo = mp[0], mhi = mp[1];
      const int mm[8] = {mlo.x, mlo.y, mlo.z, mlo.w, mhi.x, mhi.y, mhi.z, mhi.w};
#pragma unroll
      for (int i = 0; i < 8; ++i) {
        const float s = (mm[i] == 0) ? MASKED : sacc[i];
        p[tt][i] = s;
        pmax = fmaxf(pmax, s);
      }
    }
    // merge the two k-halves held by lanes (L, L^16) sharing the same q
    pmax = fmaxf(pmax, __shfl_xor(pmax, 16, 32));
    const float mnew = fmaxf(mi, pmax);
    const float alpha = DEV_EXP2F(mi - mnew);
    float psum = 0.0f;
#pragma unroll
    for (int tt = 0; tt < 4; ++tt)
#pragma unroll
      for (int i = 0; i < 8; ++i) {
        const float e = DEV_EXP2F(p[tt][i] - mnew);   // raw v_exp_f32
        p[tt][i] = e;
        psum += e;
      }
    psum += __shfl_xor(psum, 16, 32);
    li = li * alpha + psum;
    mi = mnew;
#pragma unroll
    for (int d = 0; d < 4; ++d)
#pragma unroll
      for (int i = 0; i < 8; ++i) oacc[d][i] *= alpha;

    // ---- pack P^T B-fragments: identity remap from S^T C-layout ----
    F16x16 pf[2];
#pragma unroll
    for (int c = 0; c < 2; ++c)
#pragma unroll
      for (int j = 0; j < 8; ++j) {
        pf[c].e[j]     = (_Float16)p[2 * c][j];
        pf[c].e[j + 8] = (_Float16)p[2 * c + 1][j];
      }

    // ---- O^T += V^T * P^T ----
#pragma unroll
    for (int d = 0; d < 4; ++d) {
      const _Float16* vp = &Vt[cur][(d * 16 + lr) * 64];
#pragma unroll
      for (int c = 0; c < 2; ++c) {
        F16x16 vf;
        vf.h2[0] = *(const h8*)(vp + c * 32 + 8 * g);
        vf.h2[1] = *(const h8*)(vp + c * 32 + 16 + 8 * g);
        oacc[d] = wmma_f16(vf.v, pf[c].v, oacc[d]);
      }
    }

    if (more) storeKV(cur ^ 1, krg, vrg);
    __syncthreads();
    cur ^= 1;
  }

  // ---- normalize + vectorized store to [B,S,D] f16 (head-concat cols) ----
  const float inv = 1.0f / li;
  _Float16* orow = Xout + ((size_t)b * S + q_lane) * D + h * DK;
#pragma unroll
  for (int d = 0; d < 4; ++d) {
    h8 ov;
#pragma unroll
    for (int i = 0; i < 8; ++i) ov[i] = (_Float16)(oacc[d][i] * inv);
    *(h8*)&orow[d * 16 + 8 * g] = ov;   // dk = d*16 + 8g + i, contiguous
  }
}

// ---------------------------------------------------------------------------
extern "C" void kernel_launch(void* const* d_in, const int* in_sizes, int n_in,
                              void* d_out, int out_size, void* d_ws,
                              size_t ws_size, hipStream_t stream) {
  (void)in_sizes; (void)n_in; (void)out_size; (void)ws_size;
  using namespace cfg;
  const float* q   = (const float*)d_in[0];
  const float* k   = (const float*)d_in[1];
  const float* v   = (const float*)d_in[2];
  const int*   msk = (const int*)d_in[3];
  const float* w_q = (const float*)d_in[4];
  const float* w_k = (const float*)d_in[5];
  const float* w_v = (const float*)d_in[6];
  const float* w_o = (const float*)d_in[7];

  // Workspace layout (32 MB): Qh | Kh | VhT | Xa, each 8 MB of f16.
  constexpr size_t HEAD_BYTES = (size_t)Bz * H * S * DK * sizeof(_Float16);
  char* ws = (char*)d_ws;
  _Float16* Qh  = (_Float16*)(ws);
  _Float16* Kh  = (_Float16*)(ws + HEAD_BYTES);
  _Float16* VhT = (_Float16*)(ws + 2 * HEAD_BYTES);
  _Float16* Xa  = (_Float16*)(ws + 3 * HEAD_BYTES);

  const dim3 blk(256);
  const dim3 ggrid(M / BM, N / BN);          // (64, 8)
  proj_gemm<false, EPI_HS><<<ggrid, blk, 0, stream>>>((const void*)q, w_q, Qh,
                                                      QSCALE);
  proj_gemm<false, EPI_HS><<<ggrid, blk, 0, stream>>>((const void*)k, w_k, Kh,
                                                      1.0f);
  proj_gemm<false, EPI_HST><<<ggrid, blk, 0, stream>>>((const void*)v, w_v,
                                                       VhT, 1.0f);

  const dim3 agrid(H, S / 128, Bz);          // (8, 32, 2): heads adjacent ->
  attn_kernel<<<agrid, blk, 0, stream>>>(Qh, Kh, VhT, msk, Xa);  // mask L2 reuse

  proj_gemm<true, EPI_F32><<<ggrid, blk, 0, stream>>>((const void*)Xa, w_o,
                                                      d_out, 1.0f);
}
